// Attention_16432544874778
// MI455X (gfx1250) — matrix operand order repaired and checked by reading, not compiled, not run on hardware
//
#include <hip/hip_runtime.h>
#include <hip/hip_bf16.h>
#include <math.h>

// MI455X / gfx1250, wave32, WMMA bf16 path.

typedef __attribute__((ext_vector_type(16))) __bf16 v16bf;
typedef __attribute__((ext_vector_type(8)))  __bf16 v8bf;
typedef __attribute__((ext_vector_type(8)))  float  v8f;

#define SCALE_ 0.125f  // 64^-0.5

// ---- fragment helpers -------------------------------------------------------
// A-matrix (16x32 bf16) per-lane layout: row = lane%16,
//   elems 0..15 -> K = k0 + (lane/16)*16 .. +15   (kb = k0 + (lane/16)*8)
static __device__ __forceinline__ v16bf load_a_frag(const __bf16* row, int kb) {
  v8bf lo = *(const v8bf*)(row + kb + (kb & 8));
  v8bf hi = *(const v8bf*)(row + kb + (kb & 8) + 8);
  v16bf a;
#pragma unroll
  for (int i = 0; i < 8; ++i) { a[i] = lo[i]; a[i + 8] = hi[i]; }
  return a;
}

// B-matrix (32x16 bf16) per-lane layout: col = lane%16,
//   elems 0..15 -> K = k0 + (lane/16)*16 .. +15   (contiguous when B is N-major)
static __device__ __forceinline__ v16bf load_b_frag(const __bf16* p) {
  return *(const v16bf*)(p);
}

static __device__ __forceinline__ v8f wmma_bf16(v16bf a, v16bf b, v8f c) {
  return __builtin_amdgcn_wmma_f32_16x16x32_bf16(false, a, false, b, (short)0, c,
                                                 false, false);
}

// ---- conversion kernels -----------------------------------------------------
__global__ void cvt_x_kernel(const float* __restrict__ x, __bf16* __restrict__ xb,
                             int n) {
  int i = blockIdx.x * blockDim.x + threadIdx.x;
  int stride = gridDim.x * blockDim.x;
  for (; i < n; i += stride) xb[i] = (__bf16)x[i];
}

// w: [rows][cols] f32 -> wt: [cols][rows] bf16  (N-major for B-fragments)
__global__ void cvt_wt_kernel(const float* __restrict__ w, __bf16* __restrict__ wt,
                              int rows, int cols) {
  int i = blockIdx.x * blockDim.x + threadIdx.x;
  int n = rows * cols;
  int stride = gridDim.x * blockDim.x;
  for (; i < n; i += stride) {
    int r = i / cols, c = i - r * cols;
    wt[c * rows + r] = (__bf16)w[i];
  }
}

// ---- QKV projection: X[4096][512] @ Wqkv -> Q/K [16][2048][64], Vt [16][64][2048]
__global__ void qkv_gemm_kernel(const __bf16* __restrict__ xb,   // [4096][512]
                                const __bf16* __restrict__ wt,   // [1536][512]
                                __bf16* __restrict__ qb,
                                __bf16* __restrict__ kbuf,
                                __bf16* __restrict__ vt) {
  const int lane = threadIdx.x & 31;
  const int wave = threadIdx.x >> 5;
  const int wid  = blockIdx.x * 8 + wave;   // 0..6143
  const int mt   = wid & 255;               // M tile (256 of them)
  const int ns   = wid >> 8;                // N supertile (24 of them, 64 cols)
  const int half = lane >> 4;
  const int lidx = lane & 15;

  const __bf16* arow = xb + (size_t)(mt * 16 + lidx) * 512;
  v8f acc[4] = {};
  for (int k0 = 0; k0 < 512; k0 += 32) {
    v16bf a = load_a_frag(arow, k0 + half * 8);
#pragma unroll
    for (int t = 0; t < 4; ++t) {
      const int col = ns * 64 + t * 16 + lidx;
      v16bf b = load_b_frag(wt + (size_t)col * 512 + k0 + half * 16);
      acc[t] = wmma_bf16(a, b, acc[t]);
    }
  }
  // scatter into head-major Q/K and transposed V
#pragma unroll
  for (int t = 0; t < 4; ++t) {
    const int col = ns * 64 + t * 16 + lidx;   // 0..1535
    const int sec = col / 512;                 // 0=Q 1=K 2=V
    const int e   = col - sec * 512;
    const int hh  = e >> 6;
    const int dc  = e & 63;
#pragma unroll
    for (int i = 0; i < 8; ++i) {
      const int row = mt * 16 + i + half * 8;  // 0..4095
      const int bb  = row >> 11;
      const int nr  = row & 2047;
      const int bh  = bb * 8 + hh;
      __bf16 val = (__bf16)acc[t][i];
      if (sec == 0)      qb  [((size_t)bh * 2048 + nr) * 64 + dc] = val;
      else if (sec == 1) kbuf[((size_t)bh * 2048 + nr) * 64 + dc] = val;
      else               vt  [((size_t)bh * 64 + dc) * 2048 + nr] = val;
    }
  }
}

// ---- flash attention: one wave = one (b,h) x 16-query tile ------------------
__global__ void attn_kernel(const __bf16* __restrict__ qb,
                            const __bf16* __restrict__ kbuf,
                            const __bf16* __restrict__ vt,
                            const unsigned char* __restrict__ maskp,
                            __bf16* __restrict__ aout) {   // [4096][512]
  __shared__ __bf16 p_lds[8][16][32];
  const int lane = threadIdx.x & 31;
  const int wave = threadIdx.x >> 5;
  const int wid  = blockIdx.x * 8 + wave;   // 0..2047
  const int qt   = wid & 127;
  const int bh   = wid >> 7;                // 0..15
  const int bb   = bh >> 3;
  const int hh   = bh & 7;
  const int half = lane >> 4;
  const int lidx = lane & 15;
  const int qbase = qt * 16;

  const __bf16* qrow = qb + ((size_t)bh * 2048 + (qbase + lidx)) * 64;
  v16bf qf0 = load_a_frag(qrow, half * 8);        // dh 0..31
  v16bf qf1 = load_a_frag(qrow, 32 + half * 8);   // dh 32..63

  float m[8], l[8];
  v8f o[4] = {};
#pragma unroll
  for (int i = 0; i < 8; ++i) { m[i] = -1e30f; l[i] = 0.f; }

  const unsigned char* mrow = maskp + (size_t)bb * 2048;

  for (int k0 = 0; k0 < qbase + 16; k0 += 32) {
    // S = Q Kt for two 16-key tiles (K row-major == B N-major layout)
    v8f s0 = {}, s1 = {};
    int kr0 = k0 + lidx;      if (kr0 > 2047) kr0 = 2047;
    int kr1 = k0 + 16 + lidx; if (kr1 > 2047) kr1 = 2047;
    const __bf16* kp0 = kbuf + ((size_t)bh * 2048 + kr0) * 64;
    const __bf16* kp1 = kbuf + ((size_t)bh * 2048 + kr1) * 64;
    s0 = wmma_bf16(qf0, load_b_frag(kp0 + half * 16), s0);
    s0 = wmma_bf16(qf1, load_b_frag(kp0 + 32 + half * 16), s0);
    s1 = wmma_bf16(qf0, load_b_frag(kp1 + half * 16), s1);
    s1 = wmma_bf16(qf1, load_b_frag(kp1 + 32 + half * 16), s1);

    const int key0 = k0 + lidx;
    const int key1 = k0 + 16 + lidx;
    const bool ok0 = (key0 < 2048) && (mrow[key0 < 2048 ? key0 : 2047] != 0);
    const bool ok1 = (key1 < 2048) && (mrow[key1 < 2048 ? key1 : 2047] != 0);

#pragma unroll
    for (int i = 0; i < 8; ++i) {
      const int q_idx = qbase + i + half * 8;
      float v0 = s0[i] * SCALE_;
      float v1 = s1[i] * SCALE_;
      if (!ok0 || key0 > q_idx) v0 = -1e30f;
      if (!ok1 || key1 > q_idx) v1 = -1e30f;
      float tmax = fmaxf(v0, v1);
      for (int mm = 8; mm >= 1; mm >>= 1) tmax = fmaxf(tmax, __shfl_xor(tmax, mm));
      const float mnew  = fmaxf(m[i], tmax);
      const float alpha = __expf(m[i] - mnew);
      const float p0 = __expf(v0 - mnew);
      const float p1 = __expf(v1 - mnew);
      float rs = p0 + p1;
      for (int mm = 8; mm >= 1; mm >>= 1) rs += __shfl_xor(rs, mm);
      l[i] = l[i] * alpha + rs;
      m[i] = mnew;
#pragma unroll
      for (int t4 = 0; t4 < 4; ++t4) o[t4][i] *= alpha;
      p_lds[wave][i + half * 8][lidx]      = (__bf16)p0;   // C-layout -> LDS
      p_lds[wave][i + half * 8][16 + lidx] = (__bf16)p1;
    }
    asm volatile("s_wait_dscnt 0" ::: "memory");  // same-wave LDS transpose
    v16bf pf = load_a_frag(&p_lds[wave][lidx][0], half * 8);  // P as A 16x32
    int kbv = k0 + half * 16; if (kbv > 2048 - 16) kbv = 2048 - 16;
#pragma unroll
    for (int t4 = 0; t4 < 4; ++t4) {
      const int dc = t4 * 16 + lidx;
      o[t4] = wmma_bf16(pf, load_b_frag(vt + ((size_t)bh * 64 + dc) * 2048 + kbv),
                        o[t4]);
    }
  }

  // epilogue: O / l -> [b][n][h*dh] bf16 for the output projection
#pragma unroll
  for (int t4 = 0; t4 < 4; ++t4) {
    const int dc = t4 * 16 + lidx;
#pragma unroll
    for (int i = 0; i < 8; ++i) {
      const int q_idx = qbase + i + half * 8;
      const size_t row = (size_t)bb * 2048 + q_idx;
      aout[row * 512 + hh * 64 + dc] = (__bf16)(o[t4][i] / l[i]);
    }
  }
}

// ---- output projection: aout[4096][512] @ Wout + b --------------------------
__global__ void outproj_kernel(const __bf16* __restrict__ aout,
                               const __bf16* __restrict__ wot,  // [512][512] N-major
                               const float* __restrict__ bout,
                               float* __restrict__ out) {
  const int lane = threadIdx.x & 31;
  const int wave = threadIdx.x >> 5;
  const int wid  = blockIdx.x * 8 + wave;  // 0..2047
  const int mt   = wid & 255;
  const int ns   = wid >> 8;               // 0..7
  const int half = lane >> 4;
  const int lidx = lane & 15;

  const __bf16* arow = aout + (size_t)(mt * 16 + lidx) * 512;
  v8f acc[4] = {};
  for (int k0 = 0; k0 < 512; k0 += 32) {
    v16bf a = load_a_frag(arow, k0 + half * 8);
#pragma unroll
    for (int t = 0; t < 4; ++t) {
      const int col = ns * 64 + t * 16 + lidx;
      acc[t] = wmma_bf16(a, load_b_frag(wot + (size_t)col * 512 + k0 + half * 16),
                         acc[t]);
    }
  }
#pragma unroll
  for (int t = 0; t < 4; ++t) {
    const int col = ns * 64 + t * 16 + lidx;
    const float bv = bout[col];
#pragma unroll
    for (int i = 0; i < 8; ++i) {
      const int row = mt * 16 + i + half * 8;
      out[(size_t)row * 512 + col] = acc[t][i] + bv;
    }
  }
}

// ---- launch -----------------------------------------------------------------
extern "C" void kernel_launch(void* const* d_in, const int* in_sizes, int n_in,
                              void* d_out, int out_size, void* d_ws, size_t ws_size,
                              hipStream_t stream) {
  const float*         x    = (const float*)d_in[0];
  const unsigned char* mask = (const unsigned char*)d_in[1];
  const float*         wqkv = (const float*)d_in[2];
  const float*         wout = (const float*)d_in[3];
  const float*         bout = (const float*)d_in[4];
  float* out = (float*)d_out;

  char* ws = (char*)d_ws;
  size_t off = 0;
  auto alloc = [&](size_t bytes) -> void* {
    void* p = (void*)(ws + off);
    off += (bytes + 255) & ~size_t(255);
    return p;
  };
  __bf16* xb   = (__bf16*)alloc((size_t)4096 * 512 * 2);
  __bf16* wqt  = (__bf16*)alloc((size_t)1536 * 512 * 2);
  __bf16* wot  = (__bf16*)alloc((size_t)512 * 512 * 2);
  __bf16* qb   = (__bf16*)alloc((size_t)16 * 2048 * 64 * 2);
  __bf16* kb   = (__bf16*)alloc((size_t)16 * 2048 * 64 * 2);
  __bf16* vt   = (__bf16*)alloc((size_t)16 * 64 * 2048 * 2);
  __bf16* aout = (__bf16*)alloc((size_t)4096 * 512 * 2);

  hipLaunchKernelGGL(cvt_x_kernel,    dim3(2048), dim3(256), 0, stream, x, xb, 4096 * 512);
  hipLaunchKernelGGL(cvt_wt_kernel,   dim3(1024), dim3(256), 0, stream, wqkv, wqt, 512, 1536);
  hipLaunchKernelGGL(cvt_wt_kernel,   dim3(512),  dim3(256), 0, stream, wout, wot, 512, 512);
  hipLaunchKernelGGL(qkv_gemm_kernel, dim3(768),  dim3(256), 0, stream, xb, wqt, qb, kb, vt);
  hipLaunchKernelGGL(attn_kernel,     dim3(256),  dim3(256), 0, stream, qb, kb, vt, mask, aout);
  hipLaunchKernelGGL(outproj_kernel,  dim3(256),  dim3(256), 0, stream, aout, wot, bout, out);
}
